// NRC_MAPU_8022998908947
// MI455X (gfx1250) — compile-verified, matrix operand order checked by reading, not run on hardware
//
#include <hip/hip_runtime.h>
#include <hip/hip_bf16.h>

// ---------------------------------------------------------------------------
// CDNA5 (gfx1250) implementation of the AaD-style loss.
// Heavy work = two GEMMs vs the 100000x512 bank, done with
// v_wmma_f32_16x16x32_bf16 (bf16 inputs, f32 accumulate), fused with a
// per-row running top-6 selection so the 256x100000 / 1280x100000 distance
// matrices are never materialized in HBM.
// Round 3: fix async-builtin arg types per hipcc diagnostic (param 0 is a
// generic `int __vector(4)*`); keep pipelined WMMA loop + global_prefetch.
// ---------------------------------------------------------------------------

typedef __attribute__((ext_vector_type(16))) __bf16 v16bf;
typedef __attribute__((ext_vector_type(8)))  __bf16 v8bf;
typedef __attribute__((ext_vector_type(8)))  float  v8f;

#ifndef __has_builtin
#define __has_builtin(x) 0
#endif
#if __has_builtin(__builtin_amdgcn_global_load_async_to_lds_b128)
#define USE_ASYNC_A_STAGE 1
#else
#define USE_ASYNC_A_STAGE 0
#endif

namespace {

constexpr int BATCH  = 256;     // B
constexpr int DIM    = 512;     // D (= K of the GEMMs)
constexpr int NCLS   = 10;      // C
constexpr int NBANK  = 100000;  // N
constexpr int KNN    = 5;       // K of the reference

constexpr int NCHUNK   = 512;                              // columns per block
constexpr int NBLK     = (NBANK + NCHUNK - 1) / NCHUNK;    // 196
constexpr int SUBTILES = NCHUNK / 16;                      // 32

// GEMM block geometry: 128 threads = 4 waves; M-block = 128 rows (8 m-tiles,
// 2 per wave); 16 columns per subtile.
constexpr int MBLOCK   = 128;

// LDS layout (bytes). Row stride padded by 16B so the 16-lane x 16B ds_loads
// spread across all 64 banks.
constexpr int LDS_STRIDE = DIM * 2 + 16;                 // 1040
constexpr int A_BYTES    = MBLOCK * LDS_STRIDE;          // 133120
constexpr int B_BYTES    = 16 * LDS_STRIDE;              // 16640
constexpr int D_STRIDE   = 17;                           // padded f32 stride
constexpr int D_BYTES    = MBLOCK * D_STRIDE * 4;        // 8704
constexpr int GEMM_SMEM  = A_BYTES + B_BYTES + D_BYTES;  // 158464 (< 160KB)

struct alignas(8) TopPair { float v; int i; };

// workspace layout (bytes)
constexpr size_t OFF_FNORM = 0;                                          // 256x512 f32
constexpr size_t OFF_FNBF  = OFF_FNORM + (size_t)BATCH * DIM * 4;        // 256x512 bf16
constexpr size_t OFF_SOFT  = OFF_FNBF  + (size_t)BATCH * DIM * 2;        // 256x10 f32
constexpr size_t OFF_A2    = OFF_SOFT  + 16384;                          // 1280x512 bf16
constexpr size_t OFF_P1    = OFF_A2    + (size_t)BATCH * KNN * DIM * 2;  // 256xNBLKx6 pairs
constexpr size_t OFF_NEAR  = OFF_P1    + (size_t)BATCH * NBLK * 6 * sizeof(TopPair);
constexpr size_t OFF_P2    = OFF_NEAR  + 8192;                           // 1280xNBLKx6 pairs
constexpr size_t OFF_NN    = OFF_P2    + (size_t)BATCH * KNN * NBLK * 6 * sizeof(TopPair);

__device__ __forceinline__ unsigned short f2bf(float x) {
  unsigned int u = __float_as_uint(x);
  u += 0x7FFFu + ((u >> 16) & 1u);      // round-to-nearest-even
  return (unsigned short)(u >> 16);
}

// Sorted (descending) running top-6 insert; strict '>' so earlier-seen
// (smaller-index) candidates win ties, matching jax top_k ordering.
__device__ __forceinline__ void top6_insert(float v, int id,
                                            float (&tv)[6], int (&ti)[6]) {
  if (v <= tv[5]) return;
  if (v <= tv[4]) { tv[5] = v; ti[5] = id; return; }
  tv[5] = tv[4]; ti[5] = ti[4];
  if (v <= tv[3]) { tv[4] = v; ti[4] = id; return; }
  tv[4] = tv[3]; ti[4] = ti[3];
  if (v <= tv[2]) { tv[3] = v; ti[3] = id; return; }
  tv[3] = tv[2]; ti[3] = ti[2];
  if (v <= tv[1]) { tv[2] = v; ti[2] = id; return; }
  tv[2] = tv[1]; ti[2] = ti[1];
  if (v <= tv[0]) { tv[1] = v; ti[1] = id; return; }
  tv[1] = tv[0]; ti[1] = ti[0];
  tv[0] = v; ti[0] = id;
}

#if USE_ASYNC_A_STAGE
// hipcc says param 0 of the async builtin is `int __vector(4) *` (generic).
typedef __attribute__((vector_size(16))) int vint4;
typedef __attribute__((address_space(3))) vint4 lds_vint4;

__device__ __forceinline__ vint4* as_gv4(const void* p) {
  return (vint4*)p;   // flat pointer to global memory
}
// Flat LDS pointers carry the LDS byte offset in their low 32 bits
// (ISA 10.2); build the AS3 pointer via integers to
// dodge generic->AS3 downcast restrictions.  AS3->generic upcasts implicitly
// if the builtin wants a generic pointer here.
__device__ __forceinline__ lds_vint4* as_lv4(const void* p) {
  return (lds_vint4*)(unsigned long long)(unsigned int)(unsigned long long)p;
}
#endif

} // namespace

// ---------------------------------------------------------------------------
// Kernel 1: L2-normalize feature rows (f32 + bf16 copies) and softmax rows.
// These are exactly the values scattered into the banks at rows 0..255.
// ---------------------------------------------------------------------------
__global__ __launch_bounds__(128) void aad_prologue(
    const float* __restrict__ features, const float* __restrict__ preds,
    float* __restrict__ fnorm, unsigned short* __restrict__ fnbf,
    float* __restrict__ soft) {
  const int b = blockIdx.x, t = threadIdx.x;
  __shared__ float red[128];
  float4 x = *(const float4*)(features + (size_t)b * DIM + t * 4);
  red[t] = x.x * x.x + x.y * x.y + x.z * x.z + x.w * x.w;
  __syncthreads();
  for (int s = 64; s > 0; s >>= 1) {
    if (t < s) red[t] += red[t + s];
    __syncthreads();
  }
  const float rn = rsqrtf(red[0]);
  float4 y;
  y.x = x.x * rn; y.y = x.y * rn; y.z = x.z * rn; y.w = x.w * rn;
  *(float4*)(fnorm + (size_t)b * DIM + t * 4) = y;
  ushort4 p;
  p.x = f2bf(y.x); p.y = f2bf(y.y); p.z = f2bf(y.z); p.w = f2bf(y.w);
  *(ushort4*)(fnbf + (size_t)b * DIM + t * 4) = p;
  if (t == 0) {   // C = 10: scalar softmax is negligible
    float m = -__builtin_inff();
    for (int c = 0; c < NCLS; ++c) m = fmaxf(m, preds[b * NCLS + c]);
    float e[NCLS], s = 0.f;
    for (int c = 0; c < NCLS; ++c) { e[c] = expf(preds[b * NCLS + c] - m); s += e[c]; }
    const float inv = 1.f / s;
    for (int c = 0; c < NCLS; ++c) soft[b * NCLS + c] = e[c] * inv;
  }
}

// ---------------------------------------------------------------------------
// Kernel 2/4: fused GEMM (A[mblock,512]bf16 x bank[cols,512]^T) + per-row
// running top-6 over this block's 512 columns. grid = (NBLK, rows/128).
// ---------------------------------------------------------------------------
__global__ __launch_bounds__(128) void aad_gemm_topk(
    const unsigned short* __restrict__ Abf,   // rows x 512 bf16 row-major
    const float* __restrict__ bank,           // NBANK x 512 f32
    const float* __restrict__ fnorm,          // updated bank rows 0..255
    TopPair* __restrict__ partial) {
  extern __shared__ char smem[];
  const int tid = threadIdx.x;
  const int lane = tid & 31, wv = tid >> 5;
  const int lane16 = lane & 15, khalf = lane >> 4;
  const int m_off = blockIdx.y * MBLOCK;
  const int n0 = blockIdx.x * NCHUNK;

  // Stage this block's 128x512 bf16 A-slice into LDS (padded stride).
  // When available, use GLOBAL_LOAD_ASYNC_TO_LDS_B128 so the copy overlaps
  // with the first B-subtile staging below (ASYNCcnt-tracked DMA).
  {
    const unsigned short* Ab = Abf + (size_t)m_off * DIM;
#if USE_ASYNC_A_STAGE
    #pragma unroll
    for (int it = 0; it < 64; ++it) {
      const int id = it * 128 + tid;
      const int row = id >> 6, seg = id & 63;   // 64 x 16B per row
      __builtin_amdgcn_global_load_async_to_lds_b128(
          as_gv4(Ab + (size_t)row * DIM + seg * 8),
          as_lv4(smem + row * LDS_STRIDE + seg * 16), 0, 0);
    }
#else
    #pragma unroll
    for (int it = 0; it < 64; ++it) {
      const int id = it * 128 + tid;
      const int row = id >> 6, seg = id & 63;   // 64 x 16B per row
      const int4 v = *(const int4*)(Ab + (size_t)row * DIM + seg * 8);
      *(int4*)(smem + row * LDS_STRIDE + seg * 16) = v;
    }
#endif
  }

  float tv[6]; int ti[6];
  #pragma unroll
  for (int j = 0; j < 6; ++j) { tv[j] = -__builtin_inff(); ti[j] = -1; }

  char* BsB = smem + A_BYTES;
  float* DsF = (float*)(smem + A_BYTES + B_BYTES);

  for (int st = 0; st < SUBTILES; ++st) {
    const int gcol0 = n0 + st * 16;
    // Stage 16 bank columns (f32 -> bf16) into LDS; rows < 256 come from
    // f_norm (the .at[trg_idx].set(...) update), OOB columns become zero.
    // Also prefetch the next subtile's bank lines (global_prefetch_b8).
    #pragma unroll
    for (int i = 0; i < 16; ++i) {
      const int id = i * 128 + tid;
      const int col = id >> 7, k4 = id & 127;
      const int gcol = gcol0 + col;
      float4 x = make_float4(0.f, 0.f, 0.f, 0.f);
      if (gcol < BATCH)      x = *(const float4*)(fnorm + (size_t)gcol * DIM + k4 * 4);
      else if (gcol < NBANK) x = *(const float4*)(bank + (size_t)gcol * DIM + k4 * 4);
      const int pcol = gcol + 16;
      if ((k4 & 15) == 0 && pcol >= BATCH && pcol < NBANK)
        __builtin_prefetch(bank + (size_t)pcol * DIM + k4 * 4, 0, 1);
      ushort4 p;
      p.x = f2bf(x.x); p.y = f2bf(x.y); p.z = f2bf(x.z); p.w = f2bf(x.w);
      *(ushort4*)(BsB + col * LDS_STRIDE + k4 * 8) = p;
    }
#if USE_ASYNC_A_STAGE
    if (st == 0) {        // A-DMA must be complete before anyone reads As
#if __has_builtin(__builtin_amdgcn_s_wait_asynccnt)
      __builtin_amdgcn_s_wait_asynccnt(0);
#else
      asm volatile("s_wait_asynccnt 0x0" ::: "memory");
#endif
    }
#endif
    __syncthreads();

    // Each wave: 2 m-tiles x 16 cols, full K=512, 1-deep software pipeline:
    // the 6 operand ds_loads for step k+32 are issued *before* the two WMMAs
    // of step k, so the consumer wait becomes a partial s_wait_dscnt instead
    // of a full stall.
    v8f acc0 = {}; v8f acc1 = {};
    const int mt0 = wv * 2, mt1 = wv * 2 + 1;
    const char* Arow0 = smem + (mt0 * 16 + lane16) * LDS_STRIDE;
    const char* Arow1 = smem + (mt1 * 16 + lane16) * LDS_STRIDE;
    const char* Brow  = BsB + lane16 * LDS_STRIDE;
    // ISA 16-bit 16x32 operand layout: lanes 0-15 own K {0..7,16..23},
    // lanes 16-31 own K {8..15,24..31}  ->  per-lane byte base:
    const int kbase = khalf * 16;
    v8bf b0 = *(const v8bf*)(Brow  + kbase);
    v8bf b1 = *(const v8bf*)(Brow  + kbase + 32);
    v8bf p0 = *(const v8bf*)(Arow0 + kbase);
    v8bf p1 = *(const v8bf*)(Arow0 + kbase + 32);
    v8bf q0 = *(const v8bf*)(Arow1 + kbase);
    v8bf q1 = *(const v8bf*)(Arow1 + kbase + 32);
    #pragma unroll
    for (int ks = 0; ks < DIM; ks += 32) {
      v8bf nb0 = b0, nb1 = b1, np0 = p0, np1 = p1, nq0 = q0, nq1 = q1;
      if (ks + 32 < DIM) {
        const int byteK = (ks + 32) * 2 + kbase;
        nb0 = *(const v8bf*)(Brow  + byteK);
        nb1 = *(const v8bf*)(Brow  + byteK + 32);
        np0 = *(const v8bf*)(Arow0 + byteK);
        np1 = *(const v8bf*)(Arow0 + byteK + 32);
        nq0 = *(const v8bf*)(Arow1 + byteK);
        nq1 = *(const v8bf*)(Arow1 + byteK + 32);
      }
      const v16bf bb = __builtin_shufflevector(b0, b1,
          0, 1, 2, 3, 4, 5, 6, 7, 8, 9, 10, 11, 12, 13, 14, 15);
      const v16bf aa0 = __builtin_shufflevector(p0, p1,
          0, 1, 2, 3, 4, 5, 6, 7, 8, 9, 10, 11, 12, 13, 14, 15);
      acc0 = __builtin_amdgcn_wmma_f32_16x16x32_bf16(
          false, aa0, false, bb, (short)0, acc0, false, false);
      const v16bf aa1 = __builtin_shufflevector(q0, q1,
          0, 1, 2, 3, 4, 5, 6, 7, 8, 9, 10, 11, 12, 13, 14, 15);
      acc1 = __builtin_amdgcn_wmma_f32_16x16x32_bf16(
          false, aa1, false, bb, (short)0, acc1, false, false);
      b0 = nb0; b1 = nb1; p0 = np0; p1 = np1; q0 = nq0; q1 = nq1;
    }
    // C/D layout: lane(0-15)->M j, lane(16-31)->M j+8; N = lane16.
    #pragma unroll
    for (int j = 0; j < 8; ++j) {
      DsF[(mt0 * 16 + khalf * 8 + j) * D_STRIDE + lane16] = acc0[j];
      DsF[(mt1 * 16 + khalf * 8 + j) * D_STRIDE + lane16] = acc1[j];
    }
    __syncthreads();

    // Each thread owns one M-row: fold these 16 columns into running top-6.
    #pragma unroll
    for (int c = 0; c < 16; ++c) {
      const int gcol = gcol0 + c;
      if (gcol < NBANK) top6_insert(DsF[tid * D_STRIDE + c], gcol, tv, ti);
    }
    // No barrier needed: next subtile's B-stage touches Bs only and the next
    // compute (which rewrites Ds) sits behind the next __syncthreads().
  }

  TopPair* pr = partial + ((size_t)(m_off + tid) * NBLK + blockIdx.x) * 6;
  #pragma unroll
  for (int j = 0; j < 6; ++j) { TopPair p; p.v = tv[j]; p.i = ti[j]; pr[j] = p; }
}

// ---------------------------------------------------------------------------
// Kernel 3/5: merge per-block partial top-6 lists into exact top-6 per row;
// drop rank 0 (the self/neighbor match) => the reference's [:, 1:] slice.
// ---------------------------------------------------------------------------
__global__ __launch_bounds__(32) void aad_merge_topk(
    const TopPair* __restrict__ partial, int* __restrict__ outIdx) {
  const int row = blockIdx.x, lane = threadIdx.x;
  const TopPair* src = partial + (size_t)row * NBLK * 6;
  float tv[6]; int ti[6];
  #pragma unroll
  for (int j = 0; j < 6; ++j) { tv[j] = -__builtin_inff(); ti[j] = -1; }
  for (int c = lane; c < NBLK * 6; c += 32) {
    const TopPair p = src[c];
    if (p.i >= 0) top6_insert(p.v, p.i, tv, ti);
  }
  __shared__ float sv[32 * 6];
  __shared__ int   si[32 * 6];
  #pragma unroll
  for (int j = 0; j < 6; ++j) { sv[lane * 6 + j] = tv[j]; si[lane * 6 + j] = ti[j]; }
  __syncthreads();
  if (lane == 0) {
    #pragma unroll
    for (int j = 0; j < 6; ++j) { tv[j] = -__builtin_inff(); ti[j] = -1; }
    for (int c = 0; c < 32 * 6; ++c)
      if (si[c] >= 0) top6_insert(sv[c], si[c], tv, ti);
    for (int j = 0; j < 5; ++j) outIdx[(size_t)row * 5 + j] = ti[j + 1];
  }
}

// ---------------------------------------------------------------------------
// Kernel 4a: gather fea_near = updated_bank[idx_near] as bf16 rows for GEMM2.
// ---------------------------------------------------------------------------
__global__ __launch_bounds__(128) void aad_gather(
    const int* __restrict__ nearIdx, const float* __restrict__ bank,
    const float* __restrict__ fnorm, unsigned short* __restrict__ A2) {
  const int r = blockIdx.x, t = threadIdx.x;
  const int idx = nearIdx[r];
  const float* src = (idx < BATCH) ? (fnorm + (size_t)idx * DIM)
                                   : (bank + (size_t)idx * DIM);
  const float4 x = *(const float4*)(src + t * 4);
  ushort4 p;
  p.x = f2bf(x.x); p.y = f2bf(x.y); p.z = f2bf(x.z); p.w = f2bf(x.w);
  *(ushort4*)(A2 + (size_t)r * DIM + t * 4) = p;
}

// ---------------------------------------------------------------------------
// Kernel 6: KL terms + match weights + entropy regularizer -> scalar loss.
// ---------------------------------------------------------------------------
__global__ __launch_bounds__(256) void aad_final(
    const float* __restrict__ soft, const float* __restrict__ sbank,
    const int* __restrict__ nearIdx, const int* __restrict__ nnIdx,
    const int* __restrict__ trg, float* __restrict__ out) {
  const int b = threadIdx.x;
  __shared__ float red[256];
  __shared__ float ms[16];
  float sob[NCLS];
  #pragma unroll
  for (int c = 0; c < NCLS; ++c) sob[c] = soft[b * NCLS + c];

  float acc_kk = 0.f;
  for (int kk = 0; kk < KNN * KNN; ++kk) {
    const int idx = nnIdx[b * (KNN * KNN) + kk];
    const float* s = (idx < BATCH) ? (soft + (size_t)idx * NCLS)
                                   : (sbank + (size_t)idx * NCLS);
    float kl = 0.f;
    #pragma unroll
    for (int c = 0; c < NCLS; ++c) {
      const float sv = s[c];
      kl += sv * (logf(sv) - sob[c]);   // reference literally subtracts softmax
    }
    acc_kk += kl;
  }
  float acc_k = 0.f;
  const int tb = trg[b];
  for (int k = 0; k < KNN; ++k) {
    const int idx = nearIdx[b * KNN + k];
    const float* s = (idx < BATCH) ? (soft + (size_t)idx * NCLS)
                                   : (sbank + (size_t)idx * NCLS);
    float kl = 0.f;
    #pragma unroll
    for (int c = 0; c < NCLS; ++c) {
      const float sv = s[c];
      kl += sv * (logf(sv) - sob[c]);
    }
    int match = 0;
    #pragma unroll
    for (int j = 0; j < KNN; ++j)
      match += (nnIdx[b * (KNN * KNN) + k * KNN + j] == tb) ? 1 : 0;
    const float w = (match > 0) ? (float)match : 0.1f;
    acc_k += kl * w;
  }
  red[b] = 0.1f * acc_kk + acc_k;   // weight_kk is the constant 0.1
  if (b < NCLS) {
    float m = 0.f;
    for (int r = 0; r < BATCH; ++r) m += soft[r * NCLS + b];
    ms[b] = m * (1.f / BATCH);
  }
  __syncthreads();
  for (int s = 128; s > 0; s >>= 1) {
    if (b < s) red[b] += red[b + s];
    __syncthreads();
  }
  if (b == 0) {
    float ent = 0.f;
    for (int c = 0; c < NCLS; ++c) ent += ms[c] * logf(ms[c] + 1e-5f);
    out[0] = red[0] * (1.f / BATCH) + ent;
  }
}

// ---------------------------------------------------------------------------
extern "C" void kernel_launch(void* const* d_in, const int* in_sizes, int n_in,
                              void* d_out, int out_size, void* d_ws, size_t ws_size,
                              hipStream_t stream) {
  (void)in_sizes; (void)n_in; (void)out_size; (void)ws_size;
  const float* features = (const float*)d_in[0];
  const float* preds    = (const float*)d_in[1];
  const float* bank     = (const float*)d_in[2];
  const float* sbank    = (const float*)d_in[3];
  const int*   trg      = (const int*)d_in[4];
  float* out = (float*)d_out;

  char* ws = (char*)d_ws;
  float*          fnorm   = (float*)(ws + OFF_FNORM);
  unsigned short* fnbf    = (unsigned short*)(ws + OFF_FNBF);
  float*          soft    = (float*)(ws + OFF_SOFT);
  unsigned short* A2      = (unsigned short*)(ws + OFF_A2);
  TopPair*        p1      = (TopPair*)(ws + OFF_P1);
  int*            nearIdx = (int*)(ws + OFF_NEAR);
  TopPair*        p2      = (TopPair*)(ws + OFF_P2);
  int*            nnIdx   = (int*)(ws + OFF_NN);

  aad_prologue<<<BATCH, 128, 0, stream>>>(features, preds, fnorm, fnbf, soft);
  // distance = f_norm @ bank.T, fused top-6
  aad_gemm_topk<<<dim3(NBLK, BATCH / MBLOCK), 128, GEMM_SMEM, stream>>>(
      fnbf, bank, fnorm, p1);
  aad_merge_topk<<<BATCH, 32, 0, stream>>>(p1, nearIdx);
  aad_gather<<<BATCH * KNN, 128, 0, stream>>>(nearIdx, bank, fnorm, A2);
  // distance_ = fea_near @ bank.T, fused top-6
  aad_gemm_topk<<<dim3(NBLK, (BATCH * KNN) / MBLOCK), 128, GEMM_SMEM, stream>>>(
      A2, bank, fnorm, p2);
  aad_merge_topk<<<BATCH * KNN, 32, 0, stream>>>(p2, nnIdx);
  aad_final<<<1, 256, 0, stream>>>(soft, sbank, nearIdx, nnIdx, trg, out);
}
